// PoincareInputBlock_21199958573155
// MI455X (gfx1250) — matrix operand order repaired and checked by reading, not compiled, not run on hardware
//
#include <hip/hip_runtime.h>
#include <hip/hip_bf16.h>

// Poincare input block for MI455X (gfx1250, wave32).
// Pipeline: prep(u0) -> WMMA bf16 im2col conv(+radial scale, BN stats) -> BN+ReLU+exp epilogue.
// Workspace layout (bytes):
//   [0,512)         : BN stats (sum[64], sumsq[64]) fp32
//   [512,4608)      : packed bf16 weight fragments (4 ntiles x 32 lanes x 16 vals)
//   [8192, +13.07MB): u0 = padded tangent input [16][226][226][4] fp32
//   [13083648, +205.5MB): u1 = post-conv tangent field [pix][64] fp32

typedef __attribute__((ext_vector_type(16))) __bf16 v16bf;
typedef __attribute__((ext_vector_type(8)))  float  v8f;

#define BATCH 16
#define HH 224
#define WW 224
#define HP 226
#define WP 226
#define COUT 64
#define NPIX (BATCH * HH * WW)          // 802816
#define NTILES (NPIX / 16)              // 50176 (16 pixels per wave tile)
#define U0_OFF 8192ULL
#define U1_OFF 13083648ULL

__device__ __forceinline__ unsigned short f2bf(float f) {
    unsigned int u = __float_as_uint(f);
    unsigned int r = u + 0x7FFFu + ((u >> 16) & 1u);   // round-to-nearest-even
    return (unsigned short)(r >> 16);
}

// artanh(min(tanh(max(r,eps)),0.999))-style radial scale for expmap0 o logmap0.
__device__ __forceinline__ float exp_log_scale(float r) {
    float n  = fmaxf(r, 1e-7f);
    float e  = tanhf(n) / n;              // expmap0 factor
    float hn = e * r;                     // norm after expmap0
    float pp = fminf(1.0f, 0.999f / fmaxf(hn, 1e-7f));   // projx
    e *= pp; hn *= pp;
    float n2 = fmaxf(hn, 1e-7f);
    float t  = fminf(n2, 1.0f - 1e-5f);
    return e * atanhf(t) / n2;            // logmap0 factor folded in
}

// ---- Kernel 0: zero BN stats + pack conv weights into B-fragment layout ----
__global__ __launch_bounds__(128) void init_kernel(const float* __restrict__ w,
                                                   float* __restrict__ stats,
                                                   unsigned short* __restrict__ wpack) {
    int tid = threadIdx.x;                // 128 threads
    stats[tid] = 0.0f;
    int nt = tid >> 5, lane = tid & 31;
    int n  = nt * 16 + (lane & 15);       // output channel column
    int kb = (lane < 16) ? 0 : 16;        // B layout: lanes 0-15 K=0..15, 16-31 K=16..31
    #pragma unroll
    for (int j = 0; j < 16; ++j) {
        int K = kb + j;                   // K = (ky*3+kx)*3 + ci, HWIO flat = K*64 + co
        float v = (K < 27) ? w[K * COUT + n] : 0.0f;
        wpack[(nt * 32 + lane) * 16 + j] = f2bf(v);
    }
}

// ---- Kernel 1: NCHW -> padded NHWC tangent input: u0 = projx+logmap0(x) ----
__global__ __launch_bounds__(256) void u0_kernel(const float* __restrict__ x,
                                                 float* __restrict__ u0, int total) {
    int idx = blockIdx.x * 256 + threadIdx.x;
    if (idx >= total) return;
    int px = idx % WP; int t = idx / WP; int py = t % HP; int b = t / HP;
    float4 out = make_float4(0.f, 0.f, 0.f, 0.f);
    if (px >= 1 && px <= WW && py >= 1 && py <= HH) {
        const float* xb = x + (size_t)b * 3 * HH * WW + (size_t)(py - 1) * WW + (px - 1);
        float p0 = xb[0], p1 = xb[HH * WW], p2 = xb[2 * HH * WW];
        float n  = fmaxf(sqrtf(p0 * p0 + p1 * p1 + p2 * p2), 1e-7f);
        float s1 = fminf(1.0f, 0.999f / n);            // projx
        float np = fmaxf(n * s1, 1e-7f);
        float tt = fminf(np, 1.0f - 1e-5f);
        float s  = s1 * atanhf(tt) / np;               // logmap0
        out = make_float4(p0 * s, p1 * s, p2 * s, 0.0f);
    }
    ((float4*)u0)[idx] = out;
}

// ---- Kernel 2: WMMA conv + bias + expmap0 o logmap0 + BN-stat accumulation ----
__global__ __launch_bounds__(128) void conv_wmma_kernel(const float* __restrict__ u0,
                                                        const unsigned short* __restrict__ wpack,
                                                        const float* __restrict__ bias,
                                                        float* __restrict__ u1,
                                                        float* __restrict__ stats) {
    __shared__ __align__(16) float lsum[64];
    __shared__ __align__(16) float lsq[64];
    int tid = threadIdx.x;
    if (tid < 64) lsum[tid] = 0.0f; else lsq[tid - 64] = 0.0f;
    __syncthreads();

    int wave = tid >> 5, lane = tid & 31;
    int tile = blockIdx.x * 4 + wave;                 // grid is exact: NTILES/4 blocks
    int xt = tile % 14; int t2 = tile / 14; int y = t2 % HH; int b = t2 / HH;
    int x0 = xt * 16;
    int ml = lane & 15;                               // pixel row (A) / channel col (B,C)
    int hi = lane >> 4;

    // A fragment: 16x32 bf16, im2col row = pixel, K = 3*tap + ci (27 real, 5 zero pad)
    union { v16bf v; unsigned short s[16]; } A;
    const float* gbase = u0 + (((size_t)b * HP + y) * WP + (x0 + ml)) * 4;
    __builtin_prefetch(gbase, 0, 3);                  // global_prefetch_b8
    #pragma unroll
    for (int j = 0; j < 16; ++j) {
        int K = j + (hi << 4);                        // contiguous K order: lanes 0-15 K=0..15, 16-31 K=16..31
        float val = 0.0f;
        if (K < 27) {
            int tap = K / 3, ci = K - tap * 3;
            int ky = tap / 3, kx = tap - ky * 3;
            val = gbase[(ky * WP + kx) * 4 + ci];
        }
        A.s[j] = f2bf(val);
    }

    // B fragments (packed, one 32B load each)
    union { v16bf v; uint4 q[2]; } Bf[4];
    #pragma unroll
    for (int nt = 0; nt < 4; ++nt) {
        const uint4* p = (const uint4*)(wpack + (size_t)(nt * 32 + lane) * 16);
        Bf[nt].q[0] = p[0]; Bf[nt].q[1] = p[1];
    }

    // C init = bias broadcast (every acc element is a different row, same channel)
    v8f acc[4];
    #pragma unroll
    for (int nt = 0; nt < 4; ++nt) {
        float bz = bias[nt * 16 + ml];
        acc[nt] = (v8f){bz, bz, bz, bz, bz, bz, bz, bz};
    }
    #pragma unroll
    for (int nt = 0; nt < 4; ++nt)
        acc[nt] = __builtin_amdgcn_wmma_f32_16x16x32_bf16(
            false, A.v, false, Bf[nt].v, (short)0, acc[nt], false, false);

    // Radial scale per pixel row: reduce sum-of-squares over 64 channels.
    // Row i lives in lanes 0-15 (pixel i) / 16-31 (pixel 8+i): butterfly inside halves.
    float scl[8];
    #pragma unroll
    for (int i = 0; i < 8; ++i) {
        float ss = 0.0f;
        #pragma unroll
        for (int nt = 0; nt < 4; ++nt) { float v = acc[nt][i]; ss += v * v; }
        ss += __shfl_xor(ss, 1, 32);
        ss += __shfl_xor(ss, 2, 32);
        ss += __shfl_xor(ss, 4, 32);
        ss += __shfl_xor(ss, 8, 32);
        scl[i] = exp_log_scale(sqrtf(ss));
    }

    // Store u1 + accumulate per-channel BN statistics.
    size_t pixbase = ((size_t)b * HH + y) * WW + x0;
    #pragma unroll
    for (int nt = 0; nt < 4; ++nt) {
        int ch = nt * 16 + ml;
        float s_ = 0.0f, q_ = 0.0f;
        #pragma unroll
        for (int i = 0; i < 8; ++i) {
            int m = i + (hi << 3);
            float v = scl[i] * acc[nt][i];
            u1[(pixbase + m) * COUT + ch] = v;
            s_ += v; q_ += v * v;
        }
        atomicAdd(&lsum[ch], s_);
        atomicAdd(&lsq[ch], q_);
    }
    __syncthreads();
    if (tid < 64)       atomicAdd(&stats[tid], lsum[tid]);
    else if (tid < 128) atomicAdd(&stats[tid], lsq[tid - 64]);
}

// ---- Kernel 3: BN normalize + expmap/logmap + ReLU + expmap -> output ----
__global__ __launch_bounds__(256) void bnrelu_kernel(const float* __restrict__ u1,
                                                     const float* __restrict__ stats,
                                                     const float* __restrict__ gamma,
                                                     const float* __restrict__ beta,
                                                     float* __restrict__ out, int npix) {
    __shared__ __align__(16) float sscale[64];
    __shared__ __align__(16) float sshift[64];
    int tid = threadIdx.x;
    if (tid < 64) {
        const float invN = 1.0f / (float)NPIX;
        float mean = stats[tid] * invN;
        float var  = stats[64 + tid] * invN - mean * mean;
        float sc   = rsqrtf(var + 1e-5f) * gamma[tid];
        sscale[tid] = sc;
        sshift[tid] = beta[tid] - mean * sc;
    }
    __syncthreads();
    size_t pix = (size_t)blockIdx.x * 256 + tid;
    if (pix >= (size_t)npix) return;

    const float4* src = (const float4*)(u1 + pix * COUT);
    float4 ub[16];
    float ss = 0.0f, sp = 0.0f;
    #pragma unroll
    for (int i = 0; i < 16; ++i) {
        float4 v  = src[i];
        float4 sc = ((const float4*)sscale)[i];
        float4 sh = ((const float4*)sshift)[i];
        v.x = v.x * sc.x + sh.x; v.y = v.y * sc.y + sh.y;
        v.z = v.z * sc.z + sh.z; v.w = v.w * sc.w + sh.w;
        ub[i] = v;
        ss += v.x * v.x + v.y * v.y + v.z * v.z + v.w * v.w;
        float rx = fmaxf(v.x, 0.f), ry = fmaxf(v.y, 0.f);
        float rz = fmaxf(v.z, 0.f), rw = fmaxf(v.w, 0.f);
        sp += rx * rx + ry * ry + rz * rz + rw * rw;
    }
    float s2 = exp_log_scale(sqrtf(ss));      // expmap0 o logmap0 after BN
    float r3 = s2 * sqrtf(sp);                // norm of relu'd tangent vector
    float n3 = fmaxf(r3, 1e-7f);
    float e3 = tanhf(n3) / n3;                // final expmap0
    float hn3 = e3 * r3;
    float pp3 = fminf(1.0f, 0.999f / fmaxf(hn3, 1e-7f));
    float sf = s2 * e3 * pp3;

    float4* dst = (float4*)(out + pix * COUT);
    #pragma unroll
    for (int i = 0; i < 16; ++i) {
        float4 v = ub[i];
        dst[i] = make_float4(sf * fmaxf(v.x, 0.f), sf * fmaxf(v.y, 0.f),
                             sf * fmaxf(v.z, 0.f), sf * fmaxf(v.w, 0.f));
    }
}

extern "C" void kernel_launch(void* const* d_in, const int* in_sizes, int n_in,
                              void* d_out, int out_size, void* d_ws, size_t ws_size,
                              hipStream_t stream) {
    const float* x     = (const float*)d_in[0];
    const float* w     = (const float*)d_in[1];
    const float* bias  = (const float*)d_in[2];
    const float* gamma = (const float*)d_in[3];
    const float* beta  = (const float*)d_in[4];
    float* out = (float*)d_out;

    char* ws = (char*)d_ws;
    float*          stats = (float*)ws;
    unsigned short* wpack = (unsigned short*)(ws + 512);
    float*          u0    = (float*)(ws + U0_OFF);
    float*          u1    = (float*)(ws + U1_OFF);

    init_kernel<<<1, 128, 0, stream>>>(w, stats, wpack);

    int tot0 = BATCH * HP * WP;
    u0_kernel<<<(tot0 + 255) / 256, 256, 0, stream>>>(x, u0, tot0);

    conv_wmma_kernel<<<NTILES / 4, 128, 0, stream>>>(u0, wpack, bias, u1, stats);

    bnrelu_kernel<<<NPIX / 256, 256, 0, stream>>>(u1, stats, gamma, beta, out, NPIX);
}
